// SMHA_12283606468165
// MI455X (gfx1250) — matrix operand order repaired and checked by reading, not compiled, not run on hardware
//
#include <hip/hip_runtime.h>
#include <hip/hip_bf16.h>

// MI455X / gfx1250, wave32. All matrix math on v_wmma_f32_16x16x32_bf16.
// B=4, T=1024, D_MODEL=1024, H=16, DQK=DV=64, POS_NUM=2047 (padded to 2048).

#define TT   1024
#define HH   16
#define DD   64
#define BB   4
#define POSN 2047
#define POSP 2048

// Async global->LDS staging experiment (CDNA5 GLOBAL_LOAD_ASYNC_TO_LDS_B128).
#define USE_ASYNC_LDS 1

typedef __attribute__((ext_vector_type(16))) __bf16 v16bf;
typedef __attribute__((ext_vector_type(8)))  __bf16 v8bf;
typedef __attribute__((ext_vector_type(8)))  float  v8f;
typedef __attribute__((ext_vector_type(4)))  float  v4f;

static __device__ __forceinline__ v8f wmma_bf16(v16bf a, v16bf b, v8f c) {
  // 8 args: (neg_a, A, neg_b, B, c_mod, C, reuse_a, reuse_b)
  return __builtin_amdgcn_wmma_f32_16x16x32_bf16(false, a, false, b, (short)0, c,
                                                 false, false);
}

// A fragment (16x32 bf16): lane&15 = row M; halves hold interleaved K-chunks:
// elems 0..7 -> K = kb + half*8 + i ; elems 8..15 -> K = kb + 16 + half*8 + i
static __device__ __forceinline__ v16bf load_afrag(const __bf16* base, int stride,
                                                   int lane, int kb) {
  const int row = lane & 15, half = lane >> 4;
  const __bf16* p = base + row * stride + kb + half * 8;
  v8bf lo = *(const v8bf*)(p);
  v8bf hi = *(const v8bf*)(p + 16);
  v16bf a;
#pragma unroll
  for (int i = 0; i < 8; ++i) { a[i] = lo[i]; a[8 + i] = hi[i]; }
  return a;
}

// B fragment (32x16 bf16): lane&15 = column N; base points at B^T rows
// (row n of base == column n of B). elems 0..7 -> K = kb + half*8 + i ; elems 8..15 -> K = kb + 16 + half*8 + i
static __device__ __forceinline__ const __bf16* bfrag_addr(const __bf16* base,
                                                           int stride, int lane,
                                                           int kb) {
  return base + (size_t)(lane & 15) * stride + kb + ((lane >> 4) << 4);
}
static __device__ __forceinline__ v16bf load_bfrag(const __bf16* base, int stride,
                                                   int lane, int kb) {
  const __bf16* p = base + (size_t)(lane & 15) * stride + kb + (lane >> 4) * 8;
  v8bf lo = *(const v8bf*)(p);
  v8bf hi = *(const v8bf*)(p + 16);
  v16bf r;
#pragma unroll
  for (int i = 0; i < 8; ++i) { r[i] = lo[i]; r[8 + i] = hi[i]; }
  return r;
}

// ---------------- conversion / transposition kernel ----------------
__global__ void cvt_kernel(const float* __restrict__ posK,
                           const float* __restrict__ posV,
                           const float* __restrict__ WZ,
                           __bf16* __restrict__ posKb,
                           __bf16* __restrict__ posVTb,
                           __bf16* __restrict__ WZTb) {
  const int NK = POSP * DD;            // posK padded, row-major [2048][64]
  const int NV = DD * POSP;            // posV transposed  [64][2048] (col 2047 = 0)
  const int NW = 1024 * 1024;          // W_Z transposed   [1024][1024]
  const int total = NK + NV + NW;
  for (int idx = blockIdx.x * blockDim.x + threadIdx.x; idx < total;
       idx += gridDim.x * blockDim.x) {
    if (idx < NK) {
      int p = idx / DD, d = idx % DD;
      float v = (p < POSN) ? posK[p * DD + d] : 0.f;
      posKb[idx] = (__bf16)v;
    } else if (idx < NK + NV) {
      int j = idx - NK;
      int d = j / POSP, p = j % POSP;
      float v = (p < POSN) ? posV[p * DD + d] : 0.f;
      posVTb[j] = (__bf16)v;
    } else {
      int j = idx - NK - NV;
      int n = j >> 10, k = j & 1023;
      WZTb[j] = (__bf16)WZ[k * 1024 + n];
    }
  }
}

// ---------------- QKV projection: [64 t]x[64 d] tiles per (b,h) ----------------
__global__ __launch_bounds__(128)
void qkv_kernel(const float* __restrict__ x, const float* __restrict__ WQ,
                const float* __restrict__ WK, const float* __restrict__ WV,
                __bf16* __restrict__ Qb, __bf16* __restrict__ Kb,
                __bf16* __restrict__ Vt) {
  __shared__ __align__(32) __bf16 sX[64 * DD];
  __shared__ __align__(32) __bf16 sWT[3][DD * DD];   // transposed: [e][d]
  const int wg = blockIdx.x;
  const int tb = wg & 15;
  const int bh = wg >> 4;
  const int b = bh / HH, h = bh % HH;
  const int t0 = tb * 64;
  const int tid = threadIdx.x;

  for (int i = tid; i < 64 * DD; i += 128) {
    int t = i >> 6, d = i & 63;
    sX[i] = (__bf16)x[((size_t)(b * TT + t0 + t)) * 1024 + h * DD + d];
  }
  const float* Ws0 = WQ + h * DD * DD;
  const float* Ws1 = WK + h * DD * DD;
  const float* Ws2 = WV + h * DD * DD;
  for (int i = tid; i < 3 * DD * DD; i += 128) {
    int o = i / (DD * DD), j = i % (DD * DD);
    int e = j >> 6, d = j & 63;
    const float* w = (o == 0) ? Ws0 : (o == 1) ? Ws1 : Ws2;
    sWT[o][j] = (__bf16)w[d * DD + e];
  }
  __syncthreads();

  const int wave = tid >> 5, lane = tid & 31;
  const int lrow = lane & 15, half = lane >> 4;
  for (int tile = wave * 12; tile < wave * 12 + 12; ++tile) {
    int o = tile / 16, rem = tile % 16;
    int mt = rem >> 2, nt = rem & 3;
    v8f acc = {};
#pragma unroll
    for (int kb = 0; kb < DD; kb += 32) {
      v16bf a  = load_afrag(sX + mt * 16 * DD, DD, lane, kb);
      v16bf bm = load_bfrag(sWT[o] + nt * 16 * DD, DD, lane, kb);
      acc = wmma_bf16(a, bm, acc);
    }
    int col = nt * 16 + lrow;
    if (o == 2) {
      v8bf pk;
#pragma unroll
      for (int r = 0; r < 8; ++r) pk[r] = (__bf16)acc[r];
      *(v8bf*)(Vt + ((size_t)(bh * DD + col)) * TT + t0 + mt * 16 + half * 8) = pk;
    } else {
      __bf16* dst = (o == 0) ? Qb : Kb;
#pragma unroll
      for (int r = 0; r < 8; ++r) {
        int t = t0 + mt * 16 + half * 8 + r;
        dst[((size_t)(bh * TT + t)) * DD + col] = (__bf16)acc[r];
      }
    }
  }
}

// ---------------- attention: one (b,h, 16-row q-block) per workgroup ----------------
// LDS = 128KB arena + 32KB P = exactly 160KB -> 2 workgroups per 320KB WGP.
__global__ __launch_bounds__(128)
void attn_kernel(const __bf16* __restrict__ Qb, const __bf16* __restrict__ Kb,
                 const __bf16* __restrict__ Vt, const __bf16* __restrict__ posKb,
                 const __bf16* __restrict__ posVTb, const int* __restrict__ pos,
                 const float* __restrict__ mask, __bf16* __restrict__ ctx) {
  __shared__ __align__(32) unsigned char uw[16 * POSP * 4];    // 128 KB alias arena
  __shared__ __align__(32) __bf16 sP[16 * TT];                 // 32 KB
  __bf16* sU = (__bf16*)uw;                  // 16 x 2048 bf16 (64 KB), U matrix
  float*  sS = (float*)(uw + 16 * POSP * 2); // 16 x 1024 f32  (64 KB), raw scores
  float*  sW = (float*)uw;                   // 16 x 2048 f32  (128 KB), scatter acc

  const int wg = blockIdx.x;
  const int qblk = wg & 63;
  const int bh = wg >> 6;
  const int b = bh / HH, h = bh % HH;
  const int q0 = qblk * 16;
  const int tid = threadIdx.x, wave = tid >> 5, lane = tid & 31;
  const int lrow = lane & 15, half = lane >> 4;

  // Q block lives in registers: two A-fragments straight from global.
  const __bf16* qbase = Qb + ((size_t)(bh * TT + q0)) * DD;
  v16bf aQ0  = load_afrag(qbase, DD, lane, 0);
  v16bf aQ32 = load_afrag(qbase, DD, lane, 32);

  // Phase U: U = Q @ posK^T  ->  sU[16][2048] bf16 (WMMA)
  for (int nt = 0; nt < 32; ++nt) {
    int p0 = wave * 512 + nt * 16;
    __builtin_prefetch(bfrag_addr(posKb + (size_t)(p0 + 16) * DD, DD, lane, 0), 0, 1);
    v8f acc = {};
    v16bf b0 = load_bfrag(posKb + (size_t)p0 * DD, DD, lane, 0);
    v16bf b1 = load_bfrag(posKb + (size_t)p0 * DD, DD, lane, 32);
    acc = wmma_bf16(aQ0, b0, acc);
    acc = wmma_bf16(aQ32, b1, acc);
    int pcol = p0 + lrow;
#pragma unroll
    for (int r = 0; r < 8; ++r) sU[(half * 8 + r) * POSP + pcol] = (__bf16)acc[r];
  }

  // Phase S: raw S = Q@K^T -> sS (fix-up sweep adds pos/mask and scales)
  for (int nt = 0; nt < 16; ++nt) {
    int k0 = wave * 256 + nt * 16;
    const __bf16* kb0 = Kb + ((size_t)(bh * TT + k0)) * DD;
    __builtin_prefetch(bfrag_addr(kb0 + 16 * DD, DD, lane, 0), 0, 1);
    v8f acc = {};
    v16bf b0 = load_bfrag(kb0, DD, lane, 0);
    v16bf b1 = load_bfrag(kb0, DD, lane, 32);
    acc = wmma_bf16(aQ0, b0, acc);
    acc = wmma_bf16(aQ32, b1, acc);
    int kg = k0 + lrow;
#pragma unroll
    for (int r = 0; r < 8; ++r) sS[(half * 8 + r) * TT + kg] = acc[r];
  }
  __syncthreads();

  // Phase fix/softmax: 8 threads per row, interleaved columns (coalesced pos/mask).
  {
    int row = tid >> 3, cg = tid & 7;
    int qg = q0 + row;
    const int*   prow = pos  + (size_t)qg * TT;
    const float* mrow = mask + (size_t)qg * TT;
    float*        srow = sS + row * TT;
    const __bf16* urow = sU + row * POSP;
    __bf16*       pout = sP + row * TT;

    float mx = -3.0e38f;
    for (int c = 0; c < 128; ++c) {
      int col = cg + (c << 3);
      int p = prow[col];
      float s = (srow[col] + (float)urow[p] + mrow[col]) * 0.125f;
      srow[col] = s;
      mx = fmaxf(mx, s);
    }
    mx = fmaxf(mx, __shfl_xor(mx, 1, 32));
    mx = fmaxf(mx, __shfl_xor(mx, 2, 32));
    mx = fmaxf(mx, __shfl_xor(mx, 4, 32));

    float sum = 0.f;
    for (int c = 0; c < 128; ++c) {
      int col = cg + (c << 3);
      float e = __expf(srow[col] - mx);
      sum += e;
      pout[col] = (__bf16)e;       // unnormalized P; 1/sum folded at writeout
    }
    sum += __shfl_xor(sum, 1, 32);
    sum += __shfl_xor(sum, 2, 32);
    sum += __shfl_xor(sum, 4, 32);
    __syncthreads();               // sU / sS dead beyond this point

    // clear W, stash row sums in unused column 2047 (posVT col 2047 is zero)
    for (int i = tid; i < 16 * POSP; i += 128) sW[i] = 0.f;
    __syncthreads();
    if (cg == 0) sW[row * POSP + POSN] = sum;
    for (int c = 0; c < 128; ++c) {
      int col = cg + (c << 3);
      atomicAdd(&sW[row * POSP + prow[col]], (float)sP[row * TT + col]);
    }
  }
  __syncthreads();

  // Phase ctx: ctx = (P @ V + W @ posV) * (1/rowsum); one 16-col dv tile per wave
  {
    int d0 = wave * 16;
    const __bf16* vbase = Vt + ((size_t)(bh * DD + d0)) * TT;
    v8f acc = {};
    for (int kt = 0; kt < TT; kt += 32) {
      __builtin_prefetch(bfrag_addr(vbase, TT, lane, kt + 32), 0, 1);
      v16bf a  = load_afrag(sP, TT, lane, kt);
      v16bf bm = load_bfrag(vbase, TT, lane, kt);
      acc = wmma_bf16(a, bm, acc);
    }
    const __bf16* pvbase = posVTb + (size_t)d0 * POSP;
    for (int kt = 0; kt < POSP; kt += 32) {
      __builtin_prefetch(bfrag_addr(pvbase, POSP, lane, kt + 32), 0, 1);
      const float* wr = sW + lrow * POSP + kt + half * 8;
      v4f w0 = *(const v4f*)(wr);
      v4f w1 = *(const v4f*)(wr + 4);
      v4f w2 = *(const v4f*)(wr + 16);
      v4f w3 = *(const v4f*)(wr + 20);
      v16bf a;
#pragma unroll
      for (int i = 0; i < 4; ++i) {
        a[i]      = (__bf16)w0[i];
        a[4 + i]  = (__bf16)w1[i];
        a[8 + i]  = (__bf16)w2[i];
        a[12 + i] = (__bf16)w3[i];
      }
      v16bf bm = load_bfrag(pvbase, POSP, lane, kt);
      acc = wmma_bf16(a, bm, acc);
    }
    int dcol = d0 + lrow;
#pragma unroll
    for (int r = 0; r < 8; ++r) {
      int q = half * 8 + r;
      float inv = 1.f / sW[q * POSP + POSN];
      float v = acc[r] * inv;
      ctx[((size_t)(b * TT + q0 + q)) * 1024 + h * DD + dcol] = (__bf16)v;
    }
  }
}

// ---------------- output projection: out = ctx @ W_Z ----------------
// Stage the 64-row ctx block once in LDS (contiguous 128KB), share across waves.
// CDNA5 path: GLOBAL_LOAD_ASYNC_TO_LDS_B128 (ASYNCcnt-tracked, no VGPR roundtrip).
__global__ __launch_bounds__(128)
void out_kernel(const __bf16* __restrict__ ctxb, const __bf16* __restrict__ WZTb,
                float* __restrict__ out) {
  __shared__ __align__(32) __bf16 sA[64 * 1024];   // 128 KB -> 2 WGs per WGP
  const int wg = blockIdx.x;
  const int nb = wg & 15;    // 16 N-blocks of 64
  const int mb = wg >> 4;    // 64 M-blocks of 64  (M = B*T = 4096)
  const int tid = threadIdx.x, wave = tid >> 5, lane = tid & 31;
  const int lrow = lane & 15, half = lane >> 4;

#if USE_ASYNC_LDS
  {
    const __bf16* src = ctxb + (size_t)(mb * 64) * 1024;
    for (int i = tid; i < 64 * 1024 / 8; i += 128) {
      unsigned long long ga = (unsigned long long)(uintptr_t)(src + (size_t)i * 8);
      unsigned la = (unsigned)(uintptr_t)(sA + i * 8);   // LDS addr = low 32 bits
      asm volatile("global_load_async_to_lds_b128 %0, %1, off"
                   :: "v"(la), "v"(ga) : "memory");
    }
    asm volatile("s_wait_asynccnt 0x0" ::: "memory");
  }
#else
  {
    const v8bf* src = (const v8bf*)(ctxb + (size_t)(mb * 64) * 1024);
    v8bf* dst = (v8bf*)sA;
    for (int i = tid; i < 64 * 1024 / 8; i += 128) dst[i] = src[i];
  }
#endif
  __syncthreads();

  const int n0 = nb * 64 + wave * 16;
  const __bf16* wzbase = WZTb + (size_t)n0 * 1024;
  v8f acc0 = {}, acc1 = {}, acc2 = {}, acc3 = {};
  for (int kt = 0; kt < 1024; kt += 32) {
    __builtin_prefetch(bfrag_addr(wzbase, 1024, lane, kt + 32), 0, 1);
    v16bf bm = load_bfrag(wzbase, 1024, lane, kt);    // one B per k-step
    v16bf a0 = load_afrag(sA + 0 * 16 * 1024, 1024, lane, kt);
    acc0 = wmma_bf16(a0, bm, acc0);
    v16bf a1 = load_afrag(sA + 1 * 16 * 1024, 1024, lane, kt);
    acc1 = wmma_bf16(a1, bm, acc1);
    v16bf a2 = load_afrag(sA + 2 * 16 * 1024, 1024, lane, kt);
    acc2 = wmma_bf16(a2, bm, acc2);
    v16bf a3 = load_afrag(sA + 3 * 16 * 1024, 1024, lane, kt);
    acc3 = wmma_bf16(a3, bm, acc3);
  }
  int ncol = n0 + lrow;
#pragma unroll
  for (int r = 0; r < 8; ++r) {
    int mrow = mb * 64 + half * 8 + r;
    out[(size_t)(mrow + 0)  * 1024 + ncol] = acc0[r];
    out[(size_t)(mrow + 16) * 1024 + ncol] = acc1[r];
    out[(size_t)(mrow + 32) * 1024 + ncol] = acc2[r];
    out[(size_t)(mrow + 48) * 1024 + ncol] = acc3[r];
  }
}

extern "C" void kernel_launch(void* const* d_in, const int* in_sizes, int n_in,
                              void* d_out, int out_size, void* d_ws, size_t ws_size,
                              hipStream_t stream) {
  const float* x    = (const float*)d_in[0];
  const float* WQ   = (const float*)d_in[1];
  const float* WK   = (const float*)d_in[2];
  const float* WV   = (const float*)d_in[3];
  const float* WZ   = (const float*)d_in[4];
  const float* pK   = (const float*)d_in[5];
  const float* pV   = (const float*)d_in[6];
  const float* mask = (const float*)d_in[7];
  const int*   pos  = (const int*)d_in[8];
  float* out = (float*)d_out;

  char* ws = (char*)d_ws;
  // bf16 workspace layout (~34.8 MB total)
  __bf16* Qb     = (__bf16*)(ws + 0);                          //  8 MB [B,H,T,64]
  __bf16* Kb     = (__bf16*)(ws + (size_t)(8u << 20));         //  8 MB [B,H,T,64]
  __bf16* Vt     = (__bf16*)(ws + (size_t)(16u << 20));        //  8 MB [B,H,64,T]
  __bf16* ctx    = (__bf16*)(ws + (size_t)(24u << 20));        //  8 MB [B,T,1024]
  __bf16* posKb  = (__bf16*)(ws + (size_t)(32u << 20));        // 256 KB [2048,64]
  __bf16* posVTb = (__bf16*)(ws + (size_t)(32u << 20) + 262144);   // 256 KB [64,2048]
  __bf16* WZTb   = (__bf16*)(ws + (size_t)(32u << 20) + 524288);   //   2 MB [1024,1024]

  hipLaunchKernelGGL(cvt_kernel, dim3(512), dim3(256), 0, stream,
                     pK, pV, WZ, posKb, posVTb, WZTb);
  hipLaunchKernelGGL(qkv_kernel, dim3(BB * HH * (TT / 64)), dim3(128), 0, stream,
                     x, WQ, WK, WV, Qb, Kb, Vt);
  hipLaunchKernelGGL(attn_kernel, dim3(BB * HH * (TT / 16)), dim3(128), 0, stream,
                     Qb, Kb, Vt, posKb, posVTb, pos, mask, ctx);
  hipLaunchKernelGGL(out_kernel, dim3((BB * TT / 64) * (1024 / 64)), dim3(128), 0, stream,
                     ctx, WZTb, out);
}
